// DSFDecoder_86492051407247
// MI455X (gfx1250) — matrix operand order repaired and checked by reading, not compiled, not run on hardware
//
#include <hip/hip_runtime.h>
#include <hip/hip_bf16.h>
#include <math.h>

// ---------------------------------------------------------------------------
// DSFDecoder for MI455X (gfx1250, wave32).
// Dominant cost: conv3x3 (77 GFLOP) -> WMMA f32_16x16x32_f16 implicit GEMM,
// double-buffered LDS, weight tiles via async global->LDS DMA when available.
// Mid-pipeline activations kept in f16 (bandwidth-bound at 23.3 TB/s).
// ---------------------------------------------------------------------------

typedef _Float16 half_t;
typedef __attribute__((ext_vector_type(16))) _Float16 v16h;
typedef __attribute__((ext_vector_type(8)))  _Float16 v8h;
typedef __attribute__((ext_vector_type(8)))  float    v8f;

// ---- CDNA5 async global->LDS copy (guarded; falls back to sync LDS store) --
#if defined(__has_builtin)
#if __has_builtin(__builtin_amdgcn_global_load_async_to_lds_b128) && \
    __has_builtin(__builtin_amdgcn_s_wait_asynccnt)
#define HAS_ASYNC_LDS 1
#endif
#endif
#ifndef HAS_ASYNC_LDS
#define HAS_ASYNC_LDS 0
#endif

static __device__ __forceinline__ void async_copy16(const half_t* g, half_t* l) {
#if HAS_ASYNC_LDS
  // hipcc diagnostic shows param types: v4i __device__* (AS1), v4i __shared__* (AS3)
  typedef int v4i __attribute__((vector_size(16)));
  __builtin_amdgcn_global_load_async_to_lds_b128(
      (__attribute__((address_space(1))) v4i*)(void*)g,
      (__attribute__((address_space(3))) v4i*)(void*)l, 0, 0);
#else
  *reinterpret_cast<v8h*>(l) = *reinterpret_cast<const v8h*>(g);
#endif
}

static __device__ __forceinline__ void wait_async_lds() {
#if HAS_ASYNC_LDS
  __builtin_amdgcn_s_wait_asynccnt(0);
#endif
}

static __device__ __forceinline__ v16h make_v16(v8h lo, v8h hi) {
  v16h r;
#pragma unroll
  for (int i = 0; i < 8; ++i) { r[i] = lo[i]; r[i + 8] = hi[i]; }
  return r;
}

static __device__ __forceinline__ v8f wmma16(v16h a, v16h b, v8f c) {
  // D = A(16x32 f16) x B(32x16 f16) + C(16x16 f32)
  return __builtin_amdgcn_wmma_f32_16x16x32_f16(false, a, false, b,
                                                (short)0, c, false, false);
}

// ---------------------------------------------------------------------------
// 0) f32 -> f16 weight conversion
// ---------------------------------------------------------------------------
__global__ void k_f32_to_f16(const float* __restrict__ s, half_t* __restrict__ d, int n) {
  int i = blockIdx.x * 256 + threadIdx.x;
  if (i < n) d[i] = (half_t)s[i];
}

// ---------------------------------------------------------------------------
// 1) Dynamic kernel generation: pooled -> GELU(W1) -> W2 -> softmax(9)
// ---------------------------------------------------------------------------
__global__ void k_dyn(const float* __restrict__ x,   // [B,128,4096]
                      const float* __restrict__ w1,  // [16,128]
                      const float* __restrict__ w2,  // [1152,16]
                      float* __restrict__ wk)        // [B,128,9]
{
  __shared__ float partial[256];
  __shared__ float pooled[128];
  __shared__ float hbuf[16];
  __shared__ float logits[1152];

  const int b = blockIdx.x;
  const int t = threadIdx.x;

  {
    const int c = t >> 1, hsel = t & 1;
    const float* xp = x + ((size_t)b * 128 + c) * 4096 + hsel * 2048;
    float s = 0.f;
    for (int i = 0; i < 2048; ++i) s += xp[i];
    partial[t] = s;
  }
  __syncthreads();
  if (t < 128) pooled[t] = (partial[2 * t] + partial[2 * t + 1]) * (1.0f / 4096.0f);
  __syncthreads();

  if (t < 16) {
    float acc = 0.f;
    for (int c = 0; c < 128; ++c) acc += pooled[c] * w1[t * 128 + c];
    hbuf[t] = 0.5f * acc * (1.0f + erff(acc * 0.70710678118654752f)); // exact GELU
  }
  __syncthreads();

  for (int o = t; o < 1152; o += 256) {
    float acc = 0.f;
#pragma unroll
    for (int j = 0; j < 16; ++j) acc += hbuf[j] * w2[o * 16 + j];
    logits[o] = acc;
  }
  __syncthreads();

  if (t < 128) {
    const int base = t * 9;
    float mx = logits[base];
#pragma unroll
    for (int k = 1; k < 9; ++k) mx = fmaxf(mx, logits[base + k]);
    float e[9], s = 0.f;
#pragma unroll
    for (int k = 0; k < 9; ++k) { e[k] = expf(logits[base + k] - mx); s += e[k]; }
    const float inv = 1.0f / s;
#pragma unroll
    for (int k = 0; k < 9; ++k) wk[((size_t)b * 128 + t) * 9 + k] = e[k] * inv;
  }
}

// ---------------------------------------------------------------------------
// 2) Dynamic depthwise 3x3 + residual (+optional ReLU); optional f32/f16 out
// ---------------------------------------------------------------------------
__global__ void k_dsc(const float* __restrict__ x, const float* __restrict__ wk,
                      float* __restrict__ out32, half_t* __restrict__ out16,
                      int do_relu)
{
  const int idx = blockIdx.x;
  const int rg = idx & 15, c = (idx >> 4) & 127, b = idx >> 11;
  const int t = threadIdx.x;
  const int h = rg * 4 + (t >> 6), w = t & 63;

  const float* base = x + ((size_t)b * 128 + c) * 4096;
  float k9[9];
#pragma unroll
  for (int i = 0; i < 9; ++i) k9[i] = wk[((size_t)b * 128 + c) * 9 + i];

  float acc = 0.f;
#pragma unroll
  for (int kh = 0; kh < 3; ++kh) {
    const int hh = h + kh - 1;
    if (hh >= 0 && hh < 64) {
#pragma unroll
      for (int kw = 0; kw < 3; ++kw) {
        const int ww = w + kw - 1;
        if (ww >= 0 && ww < 64) acc += k9[kh * 3 + kw] * base[hh * 64 + ww];
      }
    }
  }
  float v = acc + base[h * 64 + w];
  if (do_relu) v = fmaxf(v, 0.f);
  const size_t oi = ((size_t)b * 128 + c) * 4096 + h * 64 + w;
  if (out32) out32[oi] = v;
  if (out16) out16[oi] = (half_t)v;
}

// ---------------------------------------------------------------------------
// 3) align 1x1: Fused[128,4096] = W[128,256] x concat(a,b)[256,4096]  (per b)
// ---------------------------------------------------------------------------
__global__ void k_align(const half_t* __restrict__ A,   // [128][256] f16
                        const half_t* __restrict__ aIn, // [B][128][4096] f16
                        const half_t* __restrict__ bIn, // [B][128][4096] f16
                        half_t* __restrict__ fused)     // [B][128][4096] f16
{
  __shared__ __align__(16) half_t bt[64][40];
  const int n0 = (blockIdx.x & 63) * 64;
  const int b = blockIdx.x >> 6;
  const int t = threadIdx.x, wid = t >> 5, lane = t & 31;
  const int l15 = lane & 15, lhi = lane >> 4;

  v8f acc[4];
#pragma unroll
  for (int j = 0; j < 4; ++j)
#pragma unroll
    for (int v = 0; v < 8; ++v) acc[j][v] = 0.f;

  for (int k0 = 0; k0 < 256; k0 += 32) {
    const int kk = t >> 3, jb = (t & 7) * 8;
    const int k = k0 + kk;
    const half_t* src = (k < 128)
        ? aIn + ((size_t)b * 128 + k) * 4096 + n0 + jb
        : bIn + ((size_t)b * 128 + (k - 128)) * 4096 + n0 + jb;
#pragma unroll
    for (int p = 0; p < 8; ++p) bt[jb + p][kk] = src[p];
    __syncthreads();

    const int m = wid * 16 + l15;
    const int kb = k0 + (lhi ? 16 : 0);
    const v8h* ap = reinterpret_cast<const v8h*>(A + m * 256 + kb);
    const v16h af = make_v16(ap[0], ap[1]);

#pragma unroll
    for (int j = 0; j < 4; ++j) {
      const int n = j * 16 + l15;
      const int kbn = lhi ? 16 : 0;
      const v8h* bp = reinterpret_cast<const v8h*>(&bt[n][kbn]);
      acc[j] = wmma16(af, make_v16(bp[0], bp[1]), acc[j]);
    }
    __syncthreads();
  }

  const int mrow = wid * 16 + (lhi ? 8 : 0);
#pragma unroll
  for (int j = 0; j < 4; ++j)
#pragma unroll
    for (int v = 0; v < 8; ++v) {
      const int m = mrow + v, n = n0 + j * 16 + l15;
      fused[((size_t)b * 128 + m) * 4096 + n] = (half_t)acc[j][v];
    }
}

// ---------------------------------------------------------------------------
// 4) conv3x3 implicit GEMM: U[512,4096] = W[512,1152] x patches  (per b)
//    Double-buffered LDS; A (weight) tiles DMA'd via async global->LDS.
// ---------------------------------------------------------------------------
__global__ void k_conv3x3(const half_t* __restrict__ W, // [512][1152] f16
                          const half_t* __restrict__ X, // [B][128][64][64] f16
                          half_t* __restrict__ U)       // [B][512][64][64] f16
{
  __shared__ __align__(16) half_t bt[2][64][40];  // B tiles, transposed+padded
  __shared__ __align__(16) half_t at[2][128][32]; // A tiles, K-major
  const int h = blockIdx.x & 63;
  const int g = (blockIdx.x >> 6) & 3;
  const int b = blockIdx.x >> 8;
  const int t = threadIdx.x, wid = t >> 5, lane = t & 31;
  const int l15 = lane & 15, lhi = lane >> 4;

  v8f acc[4];
#pragma unroll
  for (int j = 0; j < 4; ++j)
#pragma unroll
    for (int v = 0; v < 8; ++v) acc[j][v] = 0.f;

  auto stage = [&](int step, int buf) {
    const int k0 = step * 32;
    // B tile: implicit-GEMM patch row with zero-pad halo, stored [pixel][k]
    const int kk = t >> 3, k = k0 + kk;
    const int c = k / 9, r = k - c * 9;
    const int kh = r / 3, kw = r - kh * 3;
    const int hh = h + kh - 1;
    const bool rowok = (hh >= 0) && (hh < 64);
    const half_t* src = X + (((size_t)b * 128 + c) * 64 + hh) * 64;
    const int jb = (t & 7) * 8;
#pragma unroll
    for (int p = 0; p < 8; ++p) {
      const int wc = jb + p, win = wc + kw - 1;
      bt[buf][wc][kk] = (rowok && win >= 0 && win < 64) ? src[win] : (half_t)0.0f;
    }
    // A tile: 128 rows x 32 K (f16), K-contiguous -> async DMA, no transpose
#pragma unroll
    for (int q = 0; q < 2; ++q) {
      const int e = t + q * 256;
      const int m = e >> 2, ck = (e & 3) * 8;
      async_copy16(W + (size_t)(g * 128 + m) * 1152 + k0 + ck, &at[buf][m][ck]);
    }
  };

  stage(0, 0);
  wait_async_lds();
  __syncthreads();

  for (int step = 0; step < 36; ++step) {
    const int cur = step & 1;
    if (step + 1 < 36) stage(step + 1, cur ^ 1);

    const v8h* ap =
        reinterpret_cast<const v8h*>(&at[cur][wid * 16 + l15][lhi ? 16 : 0]);
    const v16h af = make_v16(ap[0], ap[1]);
#pragma unroll
    for (int j = 0; j < 4; ++j) {
      const v8h* bp =
          reinterpret_cast<const v8h*>(&bt[cur][j * 16 + l15][lhi ? 16 : 0]);
      acc[j] = wmma16(af, make_v16(bp[0], bp[1]), acc[j]);
    }
    wait_async_lds();
    __syncthreads();
  }

  const int mrow = g * 128 + wid * 16 + (lhi ? 8 : 0);
#pragma unroll
  for (int j = 0; j < 4; ++j)
#pragma unroll
    for (int v = 0; v < 8; ++v) {
      const int m = mrow + v, wcol = j * 16 + l15;
      U[((size_t)b * 512 + m) * 4096 + h * 64 + wcol] = (half_t)acc[j][v];
    }
}

// ---------------------------------------------------------------------------
// 5) pixel_shuffle fused with up_w2 1x1: one GEMM per subpixel s
// ---------------------------------------------------------------------------
__global__ void k_up(const half_t* __restrict__ W2, // [64][128] f16
                     const half_t* __restrict__ U,  // [B][512][4096] f16
                     float* __restrict__ outF,      // [B][64][128][128] f32
                     half_t* __restrict__ up16)     // [B][64][128][128] f16
{
  __shared__ __align__(16) half_t bt[128][40];
  const int nt = blockIdx.x & 31;
  const int s = (blockIdx.x >> 5) & 3;
  const int b = blockIdx.x >> 7;
  const int n0 = nt * 128;
  const int t = threadIdx.x, wid = t >> 5, lane = t & 31;
  const int l15 = lane & 15, lhi = lane >> 4;
  const int m0 = (wid & 3) * 16, nh = (wid >> 2) * 64;

  v8f acc[4];
#pragma unroll
  for (int j = 0; j < 4; ++j)
#pragma unroll
    for (int v = 0; v < 8; ++v) acc[j][v] = 0.f;

  for (int k0 = 0; k0 < 128; k0 += 32) {
    const int kk = t & 31, prow = (t >> 5) * 16;
    const int ch = 4 * (k0 + kk) + s;
    const half_t* src = U + ((size_t)b * 512 + ch) * 4096 + n0 + prow;
    if (k0 + 32 < 128)  // prefetch next K-chunk (channel +128)
      __builtin_prefetch(src + (size_t)128 * 4096, 0, 1);
#pragma unroll
    for (int p = 0; p < 16; ++p) bt[prow + p][kk] = src[p];
    __syncthreads();

    const int m = m0 + l15;
    const int kb = k0 + (lhi ? 16 : 0);
    const v8h* ap = reinterpret_cast<const v8h*>(W2 + m * 128 + kb);
    const v16h af = make_v16(ap[0], ap[1]);

#pragma unroll
    for (int j = 0; j < 4; ++j) {
      const int n = nh + j * 16 + l15;
      const int kbn = lhi ? 16 : 0;
      const v8h* bp = reinterpret_cast<const v8h*>(&bt[n][kbn]);
      acc[j] = wmma16(af, make_v16(bp[0], bp[1]), acc[j]);
    }
    __syncthreads();
  }

  const int i = s >> 1, jj = s & 1;
  const int mbase = m0 + (lhi ? 8 : 0);
#pragma unroll
  for (int j = 0; j < 4; ++j)
#pragma unroll
    for (int v = 0; v < 8; ++v) {
      const int m = mbase + v;
      const int n = n0 + nh + j * 16 + l15;
      const int hlo = n >> 6, wlo = n & 63;
      const int y = 2 * hlo + i, x2 = 2 * wlo + jj;
      const size_t oi = (((size_t)b * 64 + m) * 128 + y) * 128 + x2;
      const float val = acc[j][v];
      outF[oi] = val;
      up16[oi] = (half_t)val;
    }
}

// ---------------------------------------------------------------------------
// 6) refine conv1: 64->8 ch, 3x3, ReLU, direct LDS-tiled
// ---------------------------------------------------------------------------
__global__ void k_reconv1(const half_t* __restrict__ up16, // [B][64][128][128]
                          const float* __restrict__ w1,    // [8][64][9]
                          half_t* __restrict__ tre)        // [B][8][128][128]
{
  __shared__ half_t tile[18][18];
  __shared__ float wl[8][9];
  const int tx = blockIdx.x & 7, ty = (blockIdx.x >> 3) & 7, b = blockIdx.x >> 6;
  const int y0 = ty * 16, x0 = tx * 16;
  const int t = threadIdx.x, ly = t >> 4, lx = t & 15;

  float acc[8];
#pragma unroll
  for (int m = 0; m < 8; ++m) acc[m] = 0.f;

  for (int c = 0; c < 64; ++c) {
    for (int e = t; e < 324; e += 256) {
      const int r = e / 18, cc = e - r * 18;
      const int yy = y0 + r - 1, xx = x0 + cc - 1;
      tile[r][cc] = (yy >= 0 && yy < 128 && xx >= 0 && xx < 128)
          ? up16[(((size_t)b * 64 + c) * 128 + yy) * 128 + xx] : (half_t)0.0f;
    }
    if (t < 72) wl[t / 9][t % 9] = w1[(t / 9) * 576 + c * 9 + (t % 9)];
    __syncthreads();
#pragma unroll
    for (int m = 0; m < 8; ++m)
#pragma unroll
      for (int kh = 0; kh < 3; ++kh)
#pragma unroll
        for (int kw = 0; kw < 3; ++kw)
          acc[m] += wl[m][kh * 3 + kw] * (float)tile[ly + kh][lx + kw];
    __syncthreads();
  }
#pragma unroll
  for (int m = 0; m < 8; ++m)
    tre[(((size_t)b * 8 + m) * 128 + y0 + ly) * 128 + x0 + lx] =
        (half_t)fmaxf(acc[m], 0.f);
}

// ---------------------------------------------------------------------------
// 7) refine conv2: 8->64 ch, 3x3, fused residual add into d_out
// ---------------------------------------------------------------------------
__global__ void k_reconv2(const half_t* __restrict__ tre, // [B][8][128][128]
                          const float* __restrict__ w2,   // [64][8][9]
                          float* __restrict__ out)        // [B][64][128][128]
{
  __shared__ half_t tile[8][18][18];
  __shared__ float wl[4608];
  const int tx = blockIdx.x & 7, ty = (blockIdx.x >> 3) & 7, b = blockIdx.x >> 6;
  const int y0 = ty * 16, x0 = tx * 16;
  const int t = threadIdx.x, ly = t >> 4, lx = t & 15;

  for (int e = t; e < 2592; e += 256) {
    const int ch = e / 324, rem = e - ch * 324;
    const int r = rem / 18, cc = rem - r * 18;
    const int yy = y0 + r - 1, xx = x0 + cc - 1;
    tile[ch][r][cc] = (yy >= 0 && yy < 128 && xx >= 0 && xx < 128)
        ? tre[(((size_t)b * 8 + ch) * 128 + yy) * 128 + xx] : (half_t)0.0f;
  }
  for (int e = t; e < 4608; e += 256) wl[e] = w2[e];
  __syncthreads();

  for (int m = 0; m < 64; ++m) {
    float acc = 0.f;
#pragma unroll
    for (int c = 0; c < 8; ++c)
#pragma unroll
      for (int kh = 0; kh < 3; ++kh)
#pragma unroll
        for (int kw = 0; kw < 3; ++kw)
          acc += wl[m * 72 + c * 9 + kh * 3 + kw] *
                 (float)tile[c][ly + kh][lx + kw];
    out[(((size_t)b * 64 + m) * 128 + y0 + ly) * 128 + x0 + lx] += acc;
  }
}

// ---------------------------------------------------------------------------
// Launch: workspace layout (bytes from ws base), ~141 MB total
// ---------------------------------------------------------------------------
extern "C" void kernel_launch(void* const* d_in, const int* in_sizes, int n_in,
                              void* d_out, int out_size, void* d_ws, size_t ws_size,
                              hipStream_t stream) {
  const float* x1    = (const float*)d_in[0];
  const float* x2    = (const float*)d_in[1];
  const float* d1w1  = (const float*)d_in[2];
  const float* d1w2  = (const float*)d_in[3];
  const float* d2w1  = (const float*)d_in[4];
  const float* d2w2  = (const float*)d_in[5];
  const float* alw   = (const float*)d_in[6];
  const float* upw1  = (const float*)d_in[7];
  const float* upw2  = (const float*)d_in[8];
  const float* rew1  = (const float*)d_in[9];
  const float* rew2  = (const float*)d_in[10];
  float* outp = (float*)d_out;

  char* ws = (char*)d_ws;
  const size_t MB = 1ull << 20;
  float*  wk      = (float*)(ws + 0);          // 0.07 MB
  half_t* alw16   = (half_t*)(ws + 1 * MB);    // 64 KB
  half_t* upw116  = (half_t*)(ws + 2 * MB);    // 1.18 MB
  half_t* upw216  = (half_t*)(ws + 4 * MB);    // 16 KB
  float*  tmp     = (float*)(ws + 8 * MB);     // 33.6 MB  (dead before u16 use)
  half_t* a16     = (half_t*)(ws + 44 * MB);   // 16.8 MB  (dead before u16 use)
  half_t* b16     = (half_t*)(ws + 62 * MB);   // 16.8 MB  (dead before u16 use)
  half_t* fused16 = (half_t*)(ws + 80 * MB);   // 16.8 MB
  half_t* u16     = (half_t*)(ws + 8 * MB);    // 67.1 MB  (reuses tmp/a16/b16)
  half_t* up16    = (half_t*)(ws + 100 * MB);  // 33.6 MB
  half_t* tre     = (half_t*)(ws + 136 * MB);  // 4.2 MB

  // weights -> f16
  k_f32_to_f16<<<(32768 + 255) / 256, 256, 0, stream>>>(alw,  alw16,  32768);
  k_f32_to_f16<<<(589824 + 255) / 256, 256, 0, stream>>>(upw1, upw116, 589824);
  k_f32_to_f16<<<(8192 + 255) / 256, 256, 0, stream>>>(upw2, upw216, 8192);

  // dynamic block on x1 -> a16
  k_dyn<<<16, 256, 0, stream>>>(x1, d1w1, d1w2, wk);
  k_dsc<<<32768, 256, 0, stream>>>(x1, wk, tmp, nullptr, 1);
  k_dyn<<<16, 256, 0, stream>>>(tmp, d2w1, d2w2, wk);
  k_dsc<<<32768, 256, 0, stream>>>(tmp, wk, nullptr, a16, 0);

  // dynamic block on x2 -> b16
  k_dyn<<<16, 256, 0, stream>>>(x2, d1w1, d1w2, wk);
  k_dsc<<<32768, 256, 0, stream>>>(x2, wk, tmp, nullptr, 1);
  k_dyn<<<16, 256, 0, stream>>>(tmp, d2w1, d2w2, wk);
  k_dsc<<<32768, 256, 0, stream>>>(tmp, wk, nullptr, b16, 0);

  // align 1x1 (WMMA)
  k_align<<<16 * 64, 256, 0, stream>>>(alw16, a16, b16, fused16);

  // up conv3x3 implicit GEMM (WMMA, dominant 77 GFLOP, async LDS weights)
  k_conv3x3<<<16 * 4 * 64, 256, 0, stream>>>(upw116, fused16, u16);

  // pixel_shuffle + up_w2 1x1 (WMMA) -> d_out (f32) + up16 (f16)
  k_up<<<16 * 4 * 32, 256, 0, stream>>>(upw216, u16, outp, up16);

  // refine path, conv2 adds residual into d_out
  k_reconv1<<<16 * 64, 256, 0, stream>>>(up16, rew1, tre);
  k_reconv2<<<16 * 64, 256, 0, stream>>>(tre, rew2, outp);
}
